// GNNModel_39144331935748
// MI455X (gfx1250) — compile-verified, matrix operand order checked by reading, not run on hardware
//
#include <hip/hip_runtime.h>
#include <hip/hip_bf16.h>

typedef __attribute__((ext_vector_type(16))) _Float16 v16h;
typedef __attribute__((ext_vector_type(8)))  float    v8f;

#define C_HID    128
#define E_DIM    32
#define C_OUT    64
#define NLAYERS  4
#define LN_EPS   1e-5f

// ---------------------------------------------------------------------------
// WMMA helper: D = A(16x32 f16) * B(32x16 f16) + C (f32)
// ---------------------------------------------------------------------------
static __device__ __forceinline__ v8f wmma_f16(v16h a, v16h b, v8f c) {
  return __builtin_amdgcn_wmma_f32_16x16x32_f16(
      /*neg_a=*/false, a, /*neg_b=*/false, b,
      /*c_mod=*/(short)0, c, /*reuse_a=*/false, /*reuse_b=*/false);
}

// A-fragment (16x32, f16) from row-major f32 matrix at 32-bit element offset
// `off` = row_base + kb (kb already includes the half-wave K offset).
// Layout (ISA 7.12.2, 16-bit A): lane<16 -> K in {0..7,16..23};
// lane>=16 -> K in {8..15,24..31}. Loaded via 4x float4 (saddr+voffset+imm).
static __device__ __forceinline__ v16h load_a_frag_off(const float* __restrict__ A,
                                                       unsigned off) {
  float4 p0 = *(const float4*)(A + off);
  float4 p1 = *(const float4*)(A + off + 4);
  float4 p2 = *(const float4*)(A + off + 16);
  float4 p3 = *(const float4*)(A + off + 20);
  v16h a;
  a[0]  = (_Float16)p0.x; a[1]  = (_Float16)p0.y; a[2]  = (_Float16)p0.z; a[3]  = (_Float16)p0.w;
  a[4]  = (_Float16)p1.x; a[5]  = (_Float16)p1.y; a[6]  = (_Float16)p1.z; a[7]  = (_Float16)p1.w;
  a[8]  = (_Float16)p2.x; a[9]  = (_Float16)p2.y; a[10] = (_Float16)p2.z; a[11] = (_Float16)p2.w;
  a[12] = (_Float16)p3.x; a[13] = (_Float16)p3.y; a[14] = (_Float16)p3.z; a[15] = (_Float16)p3.w;
  return a;
}

// ---------------------------------------------------------------------------
// Weight packer: row-major f32 W[K][N] -> f16 B-fragments.
// Packed layout: frag[((nt*nk + kt)*32 + lane)*16 + h]; B-frag mirrors A layout
// with N = lane&15 striped across lanes.
// ---------------------------------------------------------------------------
__global__ void pack_b_kernel(const float* __restrict__ W, _Float16* __restrict__ dst,
                              int K, int N) {
  const int nk = K >> 5, nn = N >> 4;
  const int idx = blockIdx.x * blockDim.x + threadIdx.x;
  if (idx >= nn * nk * 32) return;
  const int lane = idx & 31;
  const int kt = (idx >> 5) % nk;
  const int nt = (idx >> 5) / nk;
  const unsigned n  = nt * 16 + (lane & 15);
  const unsigned kb = kt * 32 + ((lane >> 4) << 3);
  _Float16* out = dst + (size_t)idx * 16;
#pragma unroll
  for (int j = 0; j < 4; ++j) {
    out[2 * j]     = (_Float16)W[(kb + 2 * j) * (unsigned)N + n];
    out[2 * j + 1] = (_Float16)W[(kb + 2 * j + 1) * (unsigned)N + n];
    out[8 + 2 * j]     = (_Float16)W[(kb + 16 + 2 * j) * (unsigned)N + n];
    out[8 + 2 * j + 1] = (_Float16)W[(kb + 16 + 2 * j + 1) * (unsigned)N + n];
  }
}

// ---------------------------------------------------------------------------
// hbuf = x   (agg initialized with x so atomics produce h = x + sum(msg))
// ---------------------------------------------------------------------------
__global__ void copy_kernel(const float* __restrict__ src, float* __restrict__ dst, long n4) {
  long i = (long)blockIdx.x * blockDim.x + threadIdx.x;
  if (i < n4) ((float4*)dst)[i] = ((const float4*)src)[i];
}

// ---------------------------------------------------------------------------
// Edge kernel: one wave per 16 edges.
//   e   = edge_attr(16x32) @ We(32x128) + be        (8 WMMA, K=32 in one shot)
//   msg = relu(x[src] + e) ; atomicAdd(hbuf[dst], msg)
// All gathers/atomics use 32-bit unsigned element offsets off uniform bases so
// the backend emits saddr+voffset forms with nt*16 folded into imm offsets.
// ---------------------------------------------------------------------------
__global__ void edge_kernel(const float* __restrict__ eattr,   // [E,32]
                            const int*   __restrict__ eidx,    // [2,E]
                            const float* __restrict__ x,       // [N,128]
                            const _Float16* __restrict__ WeP,  // packed [8][1] frags
                            const float* __restrict__ be,      // [128]
                            float* __restrict__ hbuf,          // [N,128]
                            int E) {
  const int lane = threadIdx.x & 31;
  const long gw = ((long)blockIdx.x * blockDim.x + threadIdx.x) >> 5;
  if (gw * 16 >= E) return;  // wave-uniform: EXEC stays all-ones
  const unsigned e0  = (unsigned)(gw * 16);
  const unsigned hi8 = (unsigned)((lane >> 4) << 3);
  const unsigned n   = (unsigned)(lane & 15);

  // A fragment from edge_attr (offsets stay 32-bit: E*32 = 51.2M elements)
  v16h a = load_a_frag_off(eattr, (e0 + n) * (unsigned)E_DIM + hi8);

  // indices for the 8 C-rows this lane owns: edge = e0 + r + 8*(lane>=16)
  const unsigned ib = e0 + hi8;
  int4 s0 = *(const int4*)(eidx + ib);
  int4 s1 = *(const int4*)(eidx + ib + 4);
  int4 d0 = *(const int4*)(eidx + (unsigned)E + ib);
  int4 d1 = *(const int4*)(eidx + (unsigned)E + ib + 4);
  unsigned xoff[8] = {(unsigned)s0.x, (unsigned)s0.y, (unsigned)s0.z, (unsigned)s0.w,
                      (unsigned)s1.x, (unsigned)s1.y, (unsigned)s1.z, (unsigned)s1.w};
  unsigned hoff[8] = {(unsigned)d0.x, (unsigned)d0.y, (unsigned)d0.z, (unsigned)d0.w,
                      (unsigned)d1.x, (unsigned)d1.y, (unsigned)d1.z, (unsigned)d1.w};
#pragma unroll
  for (int r = 0; r < 8; ++r) {
    xoff[r] = xoff[r] * (unsigned)C_HID + n;   // one mad_u32 per row
    hoff[r] = hoff[r] * (unsigned)C_HID + n;
  }
  float bias[8];
#pragma unroll
  for (int nt = 0; nt < 8; ++nt) bias[nt] = be[nt * 16 + n];

  // all 8 WMMAs up front so they pipeline
  v8f c[8];
#pragma unroll
  for (int nt = 0; nt < 8; ++nt) {
    v16h b = *(const v16h*)(WeP + (unsigned)(nt * 32 + lane) * 16u);
    v8f z = {};
    c[nt] = wmma_f16(a, b, z);
  }

#pragma unroll
  for (int nt = 0; nt < 8; ++nt) {
    float xs[8];
#pragma unroll
    for (int r = 0; r < 8; ++r) xs[r] = x[xoff[r] + nt * 16];   // 8 gathers in flight
#pragma unroll
    for (int r = 0; r < 8; ++r) {
      float msg = fmaxf(c[nt][r] + bias[nt] + xs[r], 0.0f);
      atomicAdd(&hbuf[hoff[r] + nt * 16], msg);                 // saddr+voffset+imm
    }
  }
}

// ---------------------------------------------------------------------------
// Node MLP kernel: one wave per 16 nodes (4 waves / 128-thread block).
//   h1 = relu(h @ W1 + b1)           [16x128]x[128x256]  (16 nt x 4 kt WMMA)
//   h2 = h1 @ W2 + b2                [16x256]x[256x128]  ( 8 nt x 8 kt WMMA)
//   h2 = LayerNorm(h2)*w+b ; x_out = relu(h2 + x_res)
// LDS (8KB/wave): h1 stored directly in A-FRAGMENT order so GEMM2 A-loads are
// one contiguous 32B read per lane; region reused as f32 h2[16][128] for LN.
// ---------------------------------------------------------------------------
#define MLP_WPB 4
__global__ void mlp_kernel(const float* __restrict__ hbuf,     // [N,128] = x+agg
                           const float* __restrict__ xres,     // [N,128]
                           const _Float16* __restrict__ W1P,   // packed [16][4]
                           const float* __restrict__ b1,       // [256]
                           const _Float16* __restrict__ W2P,   // packed [8][8]
                           const float* __restrict__ b2,       // [128]
                           const float* __restrict__ lnw, const float* __restrict__ lnb,
                           float* __restrict__ xout, int Nn) {
  __shared__ __align__(16) unsigned char smem[MLP_WPB][16 * 256 * 2];
  const int lane = threadIdx.x & 31;
  const int w    = threadIdx.x >> 5;
  const long tile = (long)blockIdx.x * MLP_WPB + w;
  if (tile * 16 >= Nn) return;  // wave-uniform
  const unsigned m0 = (unsigned)(tile * 16);

  _Float16* h1f = (_Float16*)smem[w];   // A-fragment order: [(kt*32+lane)*16 + h]
  float*    h2  = (float*)smem[w];
  const unsigned hi8 = (unsigned)((lane >> 4) << 3);
  const unsigned nlo = (unsigned)(lane & 15);
  const unsigned rowoff = (m0 + nlo) * (unsigned)C_HID + hi8;  // A-frag base offset

  // ---- GEMM1: h(16x128) @ W1(128x256), relu, stash f16 frags into LDS ----
  v16h a1[4];
#pragma unroll
  for (int kt = 0; kt < 4; ++kt) a1[kt] = load_a_frag_off(hbuf, rowoff + kt * 32);

#pragma unroll
  for (int nt = 0; nt < 16; ++nt) {
    v8f c = {};
#pragma unroll
    for (int kt = 0; kt < 4; ++kt) {
      v16h b = *(const v16h*)(W1P + (unsigned)((nt * 4 + kt) * 32 + lane) * 16u);
      c = wmma_f16(a1[kt], b, c);
    }
    const float bias = b1[nt * 16 + nlo];
    // This lane owns output column ncol for rows m=r+hi8; scatter into the
    // A-fragment slot GEMM2 will read: k-tile kt2, target lane, half index hh.
    const unsigned ncol = nt * 16 + nlo;
    const unsigned kt2  = ncol >> 5;
    const unsigned kk   = ncol & 31;
    const unsigned hi   = (kk >> 3) & 1;     // upper half-wave holds K%32 in 8..15,24..31
    const unsigned kq   = kk - 8 * hi;
    const unsigned hh   = (kq < 8) ? kq : kq - 8;
    _Float16* fbase = h1f + (kt2 * 32 + 16 * hi) * 16 + hh;
#pragma unroll
    for (int r = 0; r < 8; ++r) {
      float v = fmaxf(c[r] + bias, 0.0f);
      fbase[(r + hi8) * 16] = (_Float16)v;
    }
  }

  // ---- GEMM2: h1(16x256) @ W2(256x128), A-frags are contiguous 32B LDS reads ----
  v8f c2[8];
#pragma unroll
  for (int nt = 0; nt < 8; ++nt) c2[nt] = (v8f){};
#pragma unroll
  for (int kt = 0; kt < 8; ++kt) {
    v16h a2 = *(const v16h*)(h1f + (unsigned)(kt * 32 + lane) * 16u);
#pragma unroll
    for (int nt = 0; nt < 8; ++nt) {
      v16h b = *(const v16h*)(W2P + (unsigned)((nt * 8 + kt) * 32 + lane) * 16u);
      c2[nt] = wmma_f16(a2, b, c2[nt]);
    }
  }
  // write f32 h2 over the (fully consumed) h1 region
#pragma unroll
  for (int nt = 0; nt < 8; ++nt) {
    const float bias = b2[nt * 16 + nlo];
#pragma unroll
    for (int r = 0; r < 8; ++r)
      h2[(r + hi8) * C_HID + nt * 16 + nlo] = c2[nt][r] + bias;
  }

  // ---- LayerNorm + residual + relu ----
  float4 w4 = ((const float4*)lnw)[lane];
  float4 bb4 = ((const float4*)lnb)[lane];
  for (int m = 0; m < 16; ++m) {
    float4 v = *(const float4*)(h2 + m * C_HID + lane * 4);
    float s1 = v.x + v.y + v.z + v.w;
    float s2 = v.x * v.x + v.y * v.y + v.z * v.z + v.w * v.w;
#pragma unroll
    for (int off = 16; off > 0; off >>= 1) {
      s1 += __shfl_xor(s1, off, 32);
      s2 += __shfl_xor(s2, off, 32);
    }
    const float mu = s1 * (1.0f / C_HID);
    const float var = s2 * (1.0f / C_HID) - mu * mu;
    const float rs = rsqrtf(var + LN_EPS);
    const unsigned go = (m0 + m) * (unsigned)C_HID + lane * 4;
    float4 res = *(const float4*)(xres + go);
    float4 o;
    o.x = fmaxf((v.x - mu) * rs * w4.x + bb4.x + res.x, 0.0f);
    o.y = fmaxf((v.y - mu) * rs * w4.y + bb4.y + res.y, 0.0f);
    o.z = fmaxf((v.z - mu) * rs * w4.z + bb4.z + res.z, 0.0f);
    o.w = fmaxf((v.w - mu) * rs * w4.w + bb4.w + res.w, 0.0f);
    *(float4*)(xout + go) = o;
  }
}

// ---------------------------------------------------------------------------
// Output projection: out = x @ Wout + bout   [N,128]x[128,64]
// ---------------------------------------------------------------------------
__global__ void out_kernel(const float* __restrict__ x,
                           const _Float16* __restrict__ WoP,  // packed [4][4]
                           const float* __restrict__ bout,    // [64]
                           float* __restrict__ out, int Nn) {
  const int lane = threadIdx.x & 31;
  const long tile = ((long)blockIdx.x * blockDim.x + threadIdx.x) >> 5;
  if (tile * 16 >= Nn) return;
  const unsigned m0  = (unsigned)(tile * 16);
  const unsigned hi8 = (unsigned)((lane >> 4) << 3);
  const unsigned nlo = (unsigned)(lane & 15);
  const unsigned rowoff = (m0 + nlo) * (unsigned)C_HID + hi8;

  v16h a[4];
#pragma unroll
  for (int kt = 0; kt < 4; ++kt) a[kt] = load_a_frag_off(x, rowoff + kt * 32);

#pragma unroll
  for (int nt = 0; nt < 4; ++nt) {
    v8f c = {};
#pragma unroll
    for (int kt = 0; kt < 4; ++kt) {
      v16h b = *(const v16h*)(WoP + (unsigned)((nt * 4 + kt) * 32 + lane) * 16u);
      c = wmma_f16(a[kt], b, c);
    }
    const float bias = bout[nt * 16 + nlo];
#pragma unroll
    for (int r = 0; r < 8; ++r)
      out[(m0 + r + hi8) * (unsigned)C_OUT + nt * 16 + nlo] = c[r] + bias;
  }
}

// ---------------------------------------------------------------------------
extern "C" void kernel_launch(void* const* d_in, const int* in_sizes, int n_in,
                              void* d_out, int out_size, void* d_ws, size_t ws_size,
                              hipStream_t stream) {
  const float* x0    = (const float*)d_in[0];
  const int*   eidx  = (const int*)d_in[1];
  const float* eattr = (const float*)d_in[2];
  const float* We    = (const float*)d_in[3];
  const float* be    = (const float*)d_in[4];
  const float* W1    = (const float*)d_in[5];
  const float* b1    = (const float*)d_in[6];
  const float* W2    = (const float*)d_in[7];
  const float* b2    = (const float*)d_in[8];
  const float* lnw   = (const float*)d_in[9];
  const float* lnb   = (const float*)d_in[10];
  const float* Wout  = (const float*)d_in[11];
  const float* bout  = (const float*)d_in[12];

  const int Nn = in_sizes[0] / C_HID;   // 100000
  const int E  = in_sizes[2] / E_DIM;   // 1600000

  // workspace layout
  size_t off = 0;
  auto alloc = [&](size_t bytes) {
    void* p = (char*)d_ws + off;
    off += (bytes + 255) & ~(size_t)255;
    return p;
  };
  float* xa   = (float*)alloc((size_t)Nn * C_HID * 4);
  float* xb   = (float*)alloc((size_t)Nn * C_HID * 4);
  float* hbuf = (float*)alloc((size_t)Nn * C_HID * 4);
  const size_t WeP_sz  = 8 * 1 * 32 * 16;   // halves per layer
  const size_t W1P_sz  = 16 * 4 * 32 * 16;
  const size_t W2P_sz  = 8 * 8 * 32 * 16;
  const size_t WoP_sz  = 4 * 4 * 32 * 16;
  _Float16* WeP = (_Float16*)alloc(NLAYERS * WeP_sz * 2);
  _Float16* W1P = (_Float16*)alloc(NLAYERS * W1P_sz * 2);
  _Float16* W2P = (_Float16*)alloc(NLAYERS * W2P_sz * 2);
  _Float16* WoP = (_Float16*)alloc(WoP_sz * 2);

  // pack all weights into f16 B-fragment layout
  for (int l = 0; l < NLAYERS; ++l) {
    pack_b_kernel<<<1, 256, 0, stream>>>(We + (size_t)l * E_DIM * C_HID, WeP + l * WeP_sz, E_DIM, C_HID);
    pack_b_kernel<<<8, 256, 0, stream>>>(W1 + (size_t)l * C_HID * 2 * C_HID, W1P + l * W1P_sz, C_HID, 2 * C_HID);
    pack_b_kernel<<<8, 256, 0, stream>>>(W2 + (size_t)l * 2 * C_HID * C_HID, W2P + l * W2P_sz, 2 * C_HID, C_HID);
  }
  pack_b_kernel<<<2, 256, 0, stream>>>(Wout, WoP, C_HID, C_OUT);

  const long n4 = (long)Nn * C_HID / 4;
  const int copy_blocks = (int)((n4 + 255) / 256);
  const int edge_blocks = (E / 16 + 7) / 8;                 // 8 waves/block
  const int mlp_blocks  = (Nn / 16 + MLP_WPB - 1) / MLP_WPB;

  const float* xcur = x0;
  for (int l = 0; l < NLAYERS; ++l) {
    float* xnxt = (l & 1) ? xb : xa;
    copy_kernel<<<copy_blocks, 256, 0, stream>>>(xcur, hbuf, n4);
    edge_kernel<<<edge_blocks, 256, 0, stream>>>(eattr, eidx, xcur, WeP + l * WeP_sz,
                                                 be + (size_t)l * C_HID, hbuf, E);
    mlp_kernel<<<mlp_blocks, MLP_WPB * 32, 0, stream>>>(
        hbuf, xcur, W1P + l * W1P_sz, b1 + (size_t)l * 2 * C_HID,
        W2P + l * W2P_sz, b2 + (size_t)l * C_HID,
        lnw + (size_t)l * C_HID, lnb + (size_t)l * C_HID, xnxt, Nn);
    xcur = xnxt;
  }

  const int out_blocks = (Nn / 16 + 7) / 8;  // 8 waves/block
  out_kernel<<<out_blocks, 256, 0, stream>>>(xcur, WoP, bout, (float*)d_out, Nn);
}
